// PointRendModule_15960098472389
// MI455X (gfx1250) — matrix operand order repaired and checked by reading, not compiled, not run on hardware
//
#include <hip/hip_runtime.h>

typedef __attribute__((ext_vector_type(16))) _Float16 v16h;
typedef __attribute__((ext_vector_type(8)))  _Float16 v8h;
typedef __attribute__((ext_vector_type(8)))  float    v8f;

#define BATCH   4
#define CFINE   64
#define CCOARSE 128
#define IMH     512
#define IMW     512
#define NPTS    4096
#define NSAMP   (NPTS * 3)     // 12288 oversampled
#define KSEL    3072           // beta * NPTS
#define NEXTRA  1024
#define CIN     192
#define HID     256
#define OUTC    4
#define MTOT    (BATCH * NPTS) // 16384 total points

// ---- workspace layout (bytes) ----
#define OFF_UNC   0                                   // B*NSAMP f32   = 196608 B
#define OFF_FEATS (OFF_UNC   + BATCH * NSAMP * 4)     // MTOT*CIN f16  = 6291456 B
#define OFF_W1H   (OFF_FEATS + MTOT * CIN * 2)        // 256*192 f16
#define OFF_W2H   (OFF_W1H   + HID * CIN * 2)         // 256*256 f16
#define OFF_W3H   (OFF_W2H   + HID * HID * 2)         // 4*256   f16

// ---------------------------------------------------------------------------
// bilinear sample with zero padding, coords in [0,1], align_corners=False
// base = (b*C + c) * H * W
__device__ __forceinline__ float fetch_px(const float* img, int base, int x, int y) {
    if (x < 0 || x >= IMW || y < 0 || y >= IMH) return 0.0f;
    return img[base + y * IMW + x];
}

__device__ __forceinline__ float bilinear(const float* img, int base, float px, float py) {
    float x = px * (float)IMW - 0.5f;
    float y = py * (float)IMH - 0.5f;
    float x0f = floorf(x), y0f = floorf(y);
    int   x0 = (int)x0f,   y0 = (int)y0f;
    float wx = x - x0f,    wy = y - y0f;
    float v00 = fetch_px(img, base, x0,     y0);
    float v10 = fetch_px(img, base, x0 + 1, y0);
    float v01 = fetch_px(img, base, x0,     y0 + 1);
    float v11 = fetch_px(img, base, x0 + 1, y0 + 1);
    return v00 * (1.f - wx) * (1.f - wy) + v10 * wx * (1.f - wy)
         + v01 * (1.f - wx) * wy         + v11 * wx * wy;
}

// ---------------------------------------------------------------------------
// Stage 1: uncertainty at oversampled points
__global__ void pr_uncertainty(const float* __restrict__ clogits,
                               const float* __restrict__ rpts,
                               float* __restrict__ unc) {
    int id = blockIdx.x * blockDim.x + threadIdx.x;
    if (id >= BATCH * NSAMP) return;
    int b = id / NSAMP;
    float px = rpts[id * 2 + 0];
    float py = rpts[id * 2 + 1];
    int base = (b * OUTC + 0) * IMH * IMW;    // channel 0 of coarse logits
    float v = bilinear(clogits, base, px, py);
    float sg = 1.0f / (1.0f + expf(-v));
    unc[id] = 1.0f - fabsf(sg - 0.5f) * 2.0f;
}

// ---------------------------------------------------------------------------
// Stage 2: per-batch top-KSEL selection via binary-searched threshold,
// then write points = [chosen (3072) | rand_extra (1024)] into out_pts.
__global__ void __launch_bounds__(1024)
pr_topk_points(const float* __restrict__ unc,
               const float* __restrict__ rpts,
               const float* __restrict__ rextra,
               float* __restrict__ out_pts) {
    __shared__ int s_cnt;
    const int b   = blockIdx.x;
    const int tid = threadIdx.x;
    const float* u  = unc + b * NSAMP;
    const float* rp = rpts + b * NSAMP * 2;
    float* op = out_pts + b * NPTS * 2;

    // pre-fill chosen slots (safety) and copy the random-extra tail
    for (int i = tid; i < KSEL; i += 1024) {
        op[i * 2 + 0] = rp[0];
        op[i * 2 + 1] = rp[1];
    }
    for (int j = tid; j < NEXTRA; j += 1024) {
        op[(KSEL + j) * 2 + 0] = rextra[(b * NEXTRA + j) * 2 + 0];
        op[(KSEL + j) * 2 + 1] = rextra[(b * NEXTRA + j) * 2 + 1];
    }

    // binary search threshold so that count(u >= lo) >= KSEL > count(u >= hi)
    float lo = 0.0f, hi = 1.0000001f;
    for (int it = 0; it < 24; ++it) {
        float mid = 0.5f * (lo + hi);
        if (tid == 0) s_cnt = 0;
        __syncthreads();
        int c = 0;
        for (int i = tid; i < NSAMP; i += 1024) c += (u[i] >= mid) ? 1 : 0;
        atomicAdd(&s_cnt, c);
        __syncthreads();
        int total = s_cnt;
        if (total >= KSEL) lo = mid; else hi = mid;
        __syncthreads();
    }

    // compact: definite winners first, then ties until full
    if (tid == 0) s_cnt = 0;
    __syncthreads();
    for (int i = tid; i < NSAMP; i += 1024) {
        float v = u[i];
        if (v >= hi) {
            int p = atomicAdd(&s_cnt, 1);
            if (p < KSEL) { op[p * 2 + 0] = rp[i * 2 + 0]; op[p * 2 + 1] = rp[i * 2 + 1]; }
        }
    }
    __syncthreads();
    for (int i = tid; i < NSAMP; i += 1024) {
        float v = u[i];
        if (v >= lo && v < hi) {
            int p = atomicAdd(&s_cnt, 1);
            if (p < KSEL) { op[p * 2 + 0] = rp[i * 2 + 0]; op[p * 2 + 1] = rp[i * 2 + 1]; }
        }
    }
}

// ---------------------------------------------------------------------------
// Stage 3: f32 -> f16 weight conversion
__global__ void pr_cvt_f16(const float* __restrict__ src, _Float16* __restrict__ dst, int n) {
    int id = blockIdx.x * blockDim.x + threadIdx.x;
    if (id < n) dst[id] = (_Float16)src[id];
}

// ---------------------------------------------------------------------------
// Stage 4: gather fine+coarse features at selected points -> f16 [MTOT, CIN]
__global__ void pr_sample_feats(const float* __restrict__ fine,
                                const float* __restrict__ coarse,
                                const float* __restrict__ pts,
                                _Float16* __restrict__ feats) {
    int id = blockIdx.x * blockDim.x + threadIdx.x;
    if (id >= MTOT * CIN) return;
    int m  = id / CIN;
    int ch = id - m * CIN;
    int b  = m / NPTS;
    int n  = m - b * NPTS;
    float px = pts[(b * NPTS + n) * 2 + 0];
    float py = pts[(b * NPTS + n) * 2 + 1];
    const float* img;
    int c, C;
    if (ch < CFINE) { img = fine;   c = ch;         C = CFINE;   }
    else            { img = coarse; c = ch - CFINE; C = CCOARSE; }
    int base = (b * C + c) * IMH * IMW;
    feats[m * CIN + ch] = (_Float16)bilinear(img, base, px, py);
}

// ---------------------------------------------------------------------------
// Stage 5: fused 3-layer MLP with WMMA f16 (f32 accumulate).
// One workgroup (8 waves) per tile of 16 points. Each wave owns 2 N-tiles.
__global__ void __launch_bounds__(256)
pr_mlp_fused(const _Float16* __restrict__ feats,
             const _Float16* __restrict__ W1h, const float* __restrict__ b1,
             const _Float16* __restrict__ W2h, const float* __restrict__ b2,
             const _Float16* __restrict__ W3h, const float* __restrict__ b3,
             float* __restrict__ out_logits) {
    __shared__ _Float16 lds_h1[16 * HID];
    __shared__ _Float16 lds_h2[16 * HID];

    const int tid  = threadIdx.x;
    const int lane = tid & 31;
    const int wav  = tid >> 5;
    const int m0   = blockIdx.x * 16;

    const int lrow  = lane & 15;          // M row within tile (both lane halves)
    const int alo   = (lane >> 4) * 8;    // A frag: lanes 16-31 take K+8 chunk
    const int bko   = (lane >> 4) * 8;    // B frag: lanes 16-31 take K+8 chunk
    const int mbase = (lane >> 4) * 8;    // C/D frag: lanes 16-31 hold M 8..15

    // ---- layer 1: X[16,192] @ W1^T -> H1[16,256], relu ----
    v16h a1[6];
    #pragma unroll
    for (int ks = 0; ks < 6; ++ks) {
        const _Float16* p = feats + (m0 + lrow) * CIN + ks * 32 + alo;
        v8h plo = *(const v8h*)p;
        v8h phi = *(const v8h*)(p + 16);
        v16h a;
        #pragma unroll
        for (int j = 0; j < 8; ++j) { a[j] = plo[j]; a[j + 8] = phi[j]; }
        a1[ks] = a;
    }
    #pragma unroll
    for (int t = 0; t < 2; ++t) {
        int nt = wav * 2 + t;
        int n  = nt * 16 + lrow;
        v8f acc = {};
        #pragma unroll
        for (int ks = 0; ks < 6; ++ks) {
            v16h bf = __builtin_shufflevector(*(const v8h*)(W1h + n * CIN + ks * 32 + bko), *(const v8h*)(W1h + n * CIN + ks * 32 + bko + 16), 0,1,2,3,4,5,6,7,8,9,10,11,12,13,14,15);
            acc = __builtin_amdgcn_wmma_f32_16x16x32_f16(
                false, a1[ks], false, bf, (short)0, acc, false, false);
        }
        float bias = b1[n];
        #pragma unroll
        for (int r = 0; r < 8; ++r)
            lds_h1[(mbase + r) * HID + n] = (_Float16)fmaxf(acc[r] + bias, 0.0f);
    }
    __syncthreads();

    // ---- layer 2: H1[16,256] @ W2^T -> H2[16,256], relu ----
    v16h a2[8];
    #pragma unroll
    for (int ks = 0; ks < 8; ++ks) {
        const _Float16* p = lds_h1 + lrow * HID + ks * 32 + alo;
        v8h plo = *(const v8h*)p;
        v8h phi = *(const v8h*)(p + 16);
        v16h a;
        #pragma unroll
        for (int j = 0; j < 8; ++j) { a[j] = plo[j]; a[j + 8] = phi[j]; }
        a2[ks] = a;
    }
    #pragma unroll
    for (int t = 0; t < 2; ++t) {
        int nt = wav * 2 + t;
        int n  = nt * 16 + lrow;
        v8f acc = {};
        #pragma unroll
        for (int ks = 0; ks < 8; ++ks) {
            v16h bf = __builtin_shufflevector(*(const v8h*)(W2h + n * HID + ks * 32 + bko), *(const v8h*)(W2h + n * HID + ks * 32 + bko + 16), 0,1,2,3,4,5,6,7,8,9,10,11,12,13,14,15);
            acc = __builtin_amdgcn_wmma_f32_16x16x32_f16(
                false, a2[ks], false, bf, (short)0, acc, false, false);
        }
        float bias = b2[n];
        #pragma unroll
        for (int r = 0; r < 8; ++r)
            lds_h2[(mbase + r) * HID + n] = (_Float16)fmaxf(acc[r] + bias, 0.0f);
    }
    __syncthreads();

    // ---- layer 3: H2[16,256] @ W3^T -> [16,4], scatter to [B,4,N] ----
    if (tid < 64) {
        int row = tid >> 2;
        int c   = tid & 3;
        float s = b3[c];
        const _Float16* hrow = lds_h2 + row * HID;
        const _Float16* wrow = W3h + c * HID;
        #pragma unroll 8
        for (int k = 0; k < HID; ++k) s += (float)hrow[k] * (float)wrow[k];
        int m = m0 + row;
        int b = m / NPTS;
        int n = m - b * NPTS;
        out_logits[(b * OUTC + c) * NPTS + n] = s;
    }
}

// ---------------------------------------------------------------------------
extern "C" void kernel_launch(void* const* d_in, const int* in_sizes, int n_in,
                              void* d_out, int out_size, void* d_ws, size_t ws_size,
                              hipStream_t stream) {
    const float* fine   = (const float*)d_in[0];
    const float* coarse = (const float*)d_in[1];
    const float* clog   = (const float*)d_in[2];
    const float* rpts   = (const float*)d_in[3];
    const float* rextra = (const float*)d_in[4];
    const float* W1     = (const float*)d_in[5];
    const float* b1     = (const float*)d_in[6];
    const float* W2     = (const float*)d_in[7];
    const float* b2     = (const float*)d_in[8];
    const float* W3     = (const float*)d_in[9];
    const float* b3     = (const float*)d_in[10];

    float* out_logits = (float*)d_out;                       // [B,4,NPTS]
    float* out_pts    = out_logits + BATCH * OUTC * NPTS;    // [B,NPTS,2]

    char* ws = (char*)d_ws;
    float*    unc   = (float*)(ws + OFF_UNC);
    _Float16* feats = (_Float16*)(ws + OFF_FEATS);
    _Float16* W1h   = (_Float16*)(ws + OFF_W1H);
    _Float16* W2h   = (_Float16*)(ws + OFF_W2H);
    _Float16* W3h   = (_Float16*)(ws + OFF_W3H);

    // 1) uncertainty at oversampled points
    {
        int tot = BATCH * NSAMP;
        pr_uncertainty<<<(tot + 255) / 256, 256, 0, stream>>>(clog, rpts, unc);
    }
    // 2) top-k + point concat (writes out_pts)
    pr_topk_points<<<BATCH, 1024, 0, stream>>>(unc, rpts, rextra, out_pts);
    // 3) weight conversion to f16
    pr_cvt_f16<<<(HID * CIN + 255) / 256, 256, 0, stream>>>(W1, W1h, HID * CIN);
    pr_cvt_f16<<<(HID * HID + 255) / 256, 256, 0, stream>>>(W2, W2h, HID * HID);
    pr_cvt_f16<<<(OUTC * HID + 255) / 256, 256, 0, stream>>>(W3, W3h, OUTC * HID);
    // 4) gather features at selected points
    {
        int tot = MTOT * CIN;
        pr_sample_feats<<<(tot + 255) / 256, 256, 0, stream>>>(fine, coarse, out_pts, feats);
    }
    // 5) fused WMMA MLP
    pr_mlp_fused<<<MTOT / 16, 256, 0, stream>>>(feats, W1h, b1, W2h, b2, W3h, b3, out_logits);
}
